// RNNScratch_4157528343262
// MI455X (gfx1250) — matrix operand order repaired and checked by reading, not compiled, not run on hardware
//
#include <hip/hip_runtime.h>
#include <stdint.h>

typedef _Float16 half_t;
typedef __attribute__((ext_vector_type(16))) _Float16 v16h;
typedef __attribute__((ext_vector_type(8)))  _Float16 v8h;
typedef __attribute__((ext_vector_type(4)))  _Float16 v4h;
typedef __attribute__((ext_vector_type(8)))  float    v8f;
typedef __attribute__((ext_vector_type(4)))  float    v4f;
typedef __attribute__((ext_vector_type(4)))  uint32_t u32x4;
typedef __attribute__((ext_vector_type(8)))  int      i32x8;
typedef __attribute__((ext_vector_type(4)))  int      i32x4;

// Tile configuration: workgroup computes BM x BN, K-step BK.
// 256 threads = 8 waves arranged 4(M) x 2(N); each wave: 32x64 = 2x4 WMMA tiles.
constexpr int BM  = 128;
constexpr int BN  = 128;
constexpr int BK  = 32;
constexpr int LDT = BK + 8;   // LDS row stride in halves: 80B rows (64B data + 16B pad)
constexpr int KD  = 1024;     // shared K dim (NI == NH == 1024)
constexpr int ND  = 1024;     // N (NH)

// MODE 0: xproj = A_f32 @ W + bias           (A = inputs slab, out = fp32)
// MODE 1: h     = tanh(xp + A_f16 @ W)       (A = f16 state via TDM, xp/out alias d_out[t])
template <int MODE>
__global__ __launch_bounds__(256)
void rnn_gemm_kernel(const void* __restrict__ Ap,
                     const float* __restrict__ W,
                     const float* __restrict__ biasOrXp,
                     float* __restrict__ outC,
                     void* __restrict__ stateOut)
{
    __shared__ half_t As[BM * LDT];   // A tile, row-major [m][k], 80B row pitch
    __shared__ half_t Bs[BN * LDT];   // B tile, transposed [n][k]

    const int tid  = threadIdx.x;
    const int lane = tid & 31;
    const int wave = tid >> 5;
    const int wr   = wave >> 1;        // 0..3  (M direction)
    const int wc   = wave & 1;         // 0..1  (N direction)
    const int rowBase = blockIdx.x * BM;
    const int colBase = blockIdx.y * BN;

    const int g   = lane >> 4;         // lane half-group (ISA fragment striping)
    const int l16 = lane & 15;

    // B-tile loader mapping: thread -> (n within tile, K half)
    const int bn = tid & 127;
    const int bh = tid >> 7;

    v8f acc[2][4];
#pragma unroll
    for (int mi = 0; mi < 2; ++mi)
#pragma unroll
        for (int ni = 0; ni < 4; ++ni)
#pragma unroll
            for (int r = 0; r < 8; ++r)
                acc[mi][ni][r] = 0.0f;

    // TDM descriptor pieces that do not vary with k0 (MODE 1 only).
    // D# per CDNA5 ISA:
    //   2D tile: tile_dim0=32 halves (64B rows), tile_dim1=128 rows,
    //   tensor_dim0=1024, tensor_dim1=256, dim0 stride=1024 elements,
    //   data_size=1 (2 bytes), pad: every 16 DWORDs insert 4 DWORDs (64B->80B pitch).
    const uint32_t ldsA = (uint32_t)(uintptr_t)(&As[0]);  // low 32 bits = LDS byte offset

    for (int k0 = 0; k0 < KD; k0 += BK) {
        __syncthreads();

        // ---- Stage A tile into LDS ----
        if (MODE == 0) {
            const float* A = (const float*)Ap;
#pragma unroll
            for (int i = 0; i < 4; ++i) {
                int f = tid + i * 256;            // 1024 float4 slots: 128 rows x 8 segs
                int r = f >> 3, s = f & 7;
                const float* src = A + (size_t)(rowBase + r) * KD + k0 + s * 4;
                v4f v = *(const v4f*)src;
                v4h h;
                h[0] = (half_t)v[0]; h[1] = (half_t)v[1];
                h[2] = (half_t)v[2]; h[3] = (half_t)v[3];
                *(v4h*)(&As[r * LDT + s * 4]) = h;
                __builtin_prefetch(src + BK, 0, 1);   // next k-step A tile -> global_prefetch
            }
        } else {
            // Tensor Data Mover: one wave DMAs the whole 128x32 f16 tile into LDS,
            // padding each 64B row to the 80B LDS pitch.
            if (wave == 0) {
                uint64_t gaddr = (uint64_t)(uintptr_t)Ap
                               + ((uint64_t)rowBase * KD + (uint64_t)k0) * 2u;
                u32x4 g0;
                g0[0] = 1u;                                   // count=1, no gather
                g0[1] = ldsA;                                 // lds_addr
                g0[2] = (uint32_t)(gaddr & 0xFFFFFFFFu);      // global_addr[31:0]
                g0[3] = (uint32_t)((gaddr >> 32) & 0x01FFFFFFu) | 0x80000000u; // addr[56:32] | type=2
                i32x8 g1;
                g1[0] = (int)((1u << 16)        // data_size = 2 bytes
                            | (1u << 20)        // pad_enable
                            | (3u << 22)        // pad_interval: 16 DWORDs
                            | (3u << 25));      // pad_amount: 4 DWORDs
                g1[1] = (int)(1024u << 16);     // tensor_dim0[15:0] in [63:48]
                g1[2] = (int)(256u << 16);      // tensor_dim0[31:16]=0 | tensor_dim1[15:0]
                g1[3] = (int)(32u << 16);       // tensor_dim1[31:16]=0 | tile_dim0=32
                g1[4] = 128;                    // tile_dim1=128 | tile_dim2=0
                g1[5] = 1024;                   // tensor_dim0_stride[31:0]
                g1[6] = 0;
                g1[7] = 0;
                i32x4 g2 = {0, 0, 0, 0};
                i32x4 g3 = {0, 0, 0, 0};
                asm volatile("tensor_load_to_lds %0, %1, %2, %3"
                             :: "s"(g0), "s"(g1), "s"(g2), "s"(g3)
                             : "memory");
            }
        }

        // ---- Stage B tile into LDS transposed (Bs[n][k] = W[k0+k][colBase+n]) ----
        {
            half_t tmp[16];
            const float* wp = W + (size_t)(k0 + bh * 16) * ND + colBase + bn;
#pragma unroll
            for (int kk = 0; kk < 16; ++kk)
                tmp[kk] = (half_t)wp[(size_t)kk * ND];
#pragma unroll
            for (int j = 0; j < 2; ++j)
                *(v8h*)(&Bs[bn * LDT + bh * 16 + j * 8]) = *(v8h*)(&tmp[j * 8]);
            __builtin_prefetch(wp + (size_t)BK * ND, 0, 1);  // next k-step W tile
        }

        if (MODE == 1 && wave == 0)
            __builtin_amdgcn_s_wait_tensorcnt(0);   // TDM data landed in LDS
        __syncthreads();

        // ---- Build fragments per ISA striping and issue WMMA ----
        // A frag (16x32 f16): lane (g,l16) holds K = {8g..8g+7} U {16+8g..16+8g+7} of row M=l16
        v16h afr[2];
#pragma unroll
        for (int mi = 0; mi < 2; ++mi) {
            int m = wr * 32 + mi * 16 + l16;
            v8h alo = *(const v8h*)(&As[m * LDT + 8 * g]);
            v8h ahi = *(const v8h*)(&As[m * LDT + 16 + 8 * g]);
            afr[mi] = __builtin_shufflevector(alo, ahi,
                0, 1, 2, 3, 4, 5, 6, 7, 8, 9, 10, 11, 12, 13, 14, 15);
        }
        // B frag (32x16 f16): lane (g,l16) holds K = {8g..8g+7} U {16+8g..16+8g+7} of column N=l16
        v16h bfr[4];
#pragma unroll
        for (int ni = 0; ni < 4; ++ni) {
            int n = wc * 64 + ni * 16 + l16;
            v8h blo = *(const v8h*)(&Bs[n * LDT + 8 * g]);
            v8h bhi = *(const v8h*)(&Bs[n * LDT + 16 + 8 * g]);
            bfr[ni] = __builtin_shufflevector(blo, bhi,
                0, 1, 2, 3, 4, 5, 6, 7, 8, 9, 10, 11, 12, 13, 14, 15);
        }
#pragma unroll
        for (int mi = 0; mi < 2; ++mi)
#pragma unroll
            for (int ni = 0; ni < 4; ++ni)
                acc[mi][ni] = __builtin_amdgcn_wmma_f32_16x16x32_f16(
                    false, afr[mi], false, bfr[ni],
                    (short)0, acc[mi][ni], false, false);
    }

    // ---- Epilogue ----
    // C/D layout: VGPR r, lanes 0-15 -> M=r, lanes 16-31 -> M=8+r; N = lane%16.
#pragma unroll
    for (int mi = 0; mi < 2; ++mi) {
#pragma unroll
        for (int ni = 0; ni < 4; ++ni) {
            const int row0 = rowBase + wr * 32 + mi * 16 + g * 8;
            const int col  = colBase + wc * 64 + ni * 16 + l16;
            if (MODE == 0) {
                const float b = biasOrXp[col];
#pragma unroll
                for (int r = 0; r < 8; ++r)
                    outC[(size_t)(row0 + r) * ND + col] = acc[mi][ni][r] + b;
            } else {
                half_t* so = (half_t*)stateOut;
#pragma unroll
                for (int r = 0; r < 8; ++r) {
                    size_t idx = (size_t)(row0 + r) * ND + col;
                    float v = tanhf(acc[mi][ni][r] + biasOrXp[idx]);
                    outC[idx] = v;
                    so[idx]   = (half_t)v;
                }
            }
        }
    }
}

__global__ void zero_words(uint32_t* p, int n)
{
    int i = blockIdx.x * blockDim.x + threadIdx.x;
    if (i < n) p[i] = 0u;
}

extern "C" void kernel_launch(void* const* d_in, const int* in_sizes, int n_in,
                              void* d_out, int out_size, void* d_ws, size_t ws_size,
                              hipStream_t stream)
{
    (void)in_sizes; (void)n_in; (void)out_size; (void)ws_size;

    const float* inputs = (const float*)d_in[0];   // [512, 256, 1024] fp32
    const float* W_xh   = (const float*)d_in[1];   // [1024, 1024]
    const float* W_hh   = (const float*)d_in[2];   // [1024, 1024]
    const float* b_h    = (const float*)d_in[3];   // [1, 1024]
    float* out = (float*)d_out;                    // outputs [512,256,1024] ++ state [256,1024]

    const int T = 512, Bn = 256, NH = 1024;
    const size_t stepElems = (size_t)Bn * NH;          // 262144
    char* ws  = (char*)d_ws;
    void* st0 = (void*)ws;                             // f16 state ping (512 KB)
    void* st1 = (void*)(ws + stepElems * 2);           // f16 state pong (512 KB)

    // state0 = 0 (f16): 512 KB = 131072 dwords
    zero_words<<<dim3(512), dim3(256), 0, stream>>>((uint32_t*)st0, 131072);

    // Phase 1: xproj = inputs @ W_xh + b_h  -> written straight into d_out outputs region
    {
        dim3 grid((T * Bn) / BM, NH / BN);             // (1024, 8)
        rnn_gemm_kernel<0><<<grid, 256, 0, stream>>>(inputs, W_xh, b_h, out, nullptr);
    }

    // Phase 2: serial scan, one launch per timestep (graph replay amortizes launch cost)
    {
        dim3 grid(Bn / BM, NH / BN);                   // (2, 8)
        for (int t = 0; t < T; ++t) {
            float* xp  = out + (size_t)t * stepElems;
            void* sin  = (t & 1) ? st1 : st0;
            void* sout = (t & 1) ? st0 : st1;
            rnn_gemm_kernel<1><<<grid, 256, 0, stream>>>(sin, W_hh, xp, xp, sout);
        }
    }

    // Final state output = outputs[T-1]
    hipMemcpyAsync(out + (size_t)T * stepElems,
                   out + (size_t)(T - 1) * stepElems,
                   stepElems * sizeof(float),
                   hipMemcpyDeviceToDevice, stream);
}
